// GNN_79688823210730
// MI455X (gfx1250) — matrix operand order repaired and checked by reading, not compiled, not run on hardware
//
#include <hip/hip_runtime.h>
#include <cstdint>

// ---------------------------------------------------------------------------
// GCN (2-layer + classifier) for MI455X / gfx1250.
// GEMMs use v_wmma_f32_16x16x32_bf16 (wave32, 16x16 tiles, fp32 accum).
// Edge aggregation is wave-per-edge coalesced gather + global_atomic_add_f32;
// h (51.2MB) is L2-resident on the 192MB L2, so aggregation stays on-chip.
// ---------------------------------------------------------------------------

typedef __attribute__((ext_vector_type(16))) __bf16 v16bf;
typedef __attribute__((ext_vector_type(8)))  float  v8f;

__device__ __forceinline__ __bf16 f2bf(float f) {
  unsigned int u = __builtin_bit_cast(unsigned int, f);
  u += 0x7FFFu + ((u >> 16) & 1u);              // round-to-nearest-even
  unsigned short s = (unsigned short)(u >> 16);
  return __builtin_bit_cast(__bf16, s);
}

// -------------------------------- utility ----------------------------------

__global__ __launch_bounds__(256) void k_zero_f4(float4* __restrict__ p, int n4) {
  int i = blockIdx.x * 256 + threadIdx.x;
  if (i < n4) p[i] = make_float4(0.f, 0.f, 0.f, 0.f);
}

__global__ __launch_bounds__(256) void k_deg(const int* __restrict__ dst,
                                             float* __restrict__ deg, int nedges) {
  int e = blockIdx.x * 256 + threadIdx.x;
  if (e < nedges) atomicAdd(&deg[dst[e]], 1.0f);
}

__global__ __launch_bounds__(256) void k_dinv(const float* __restrict__ deg,
                                              float* __restrict__ dinv, int n) {
  int i = blockIdx.x * 256 + threadIdx.x;
  if (i < n) dinv[i] = rsqrtf(deg[i] + 1.0f);   // +1 = self loop, always > 0
}

// ----------------------- WMMA GEMM: Y[N,128] = X[N,128] @ W[128,128] --------

__global__ __launch_bounds__(256) void k_gcn_gemm128(
    const float* __restrict__ X, const float* __restrict__ W,
    float* __restrict__ Y, int nrows) {
  __shared__ __bf16 sX [128 * 128];   // [row][k]
  __shared__ __bf16 sWt[128 * 128];   // [col][k]  (transposed for contiguous K)
  const int tid  = threadIdx.x;
  const int row0 = blockIdx.x * 128;

  // stage W transposed as bf16
  for (int idx = tid * 4; idx < 128 * 128; idx += 256 * 4) {
    const int r = idx >> 7, c = idx & 127;      // W[r=k][c=col], row-major
    const float4 w = *(const float4*)(W + idx);
    sWt[(c + 0) * 128 + r] = f2bf(w.x);
    sWt[(c + 1) * 128 + r] = f2bf(w.y);
    sWt[(c + 2) * 128 + r] = f2bf(w.z);
    sWt[(c + 3) * 128 + r] = f2bf(w.w);
  }
  // stage X tile (row-guarded, zero padded)
  for (int idx = tid * 4; idx < 128 * 128; idx += 256 * 4) {
    const int r = idx >> 7, c = idx & 127;
    const int gr = row0 + r;
    float4 x = make_float4(0.f, 0.f, 0.f, 0.f);
    if (gr < nrows) x = *(const float4*)(X + (size_t)gr * 128 + c);
    sX[idx + 0] = f2bf(x.x);
    sX[idx + 1] = f2bf(x.y);
    sX[idx + 2] = f2bf(x.z);
    sX[idx + 3] = f2bf(x.w);
  }
  __syncthreads();

  const int wave = tid >> 5, lane = tid & 31;
  const int m0   = wave * 16;                 // wave's 16-row strip
  const int mrow = m0 + (lane & 15);
  const int ka   = (lane < 16) ? 0 : 16;      // A-frag K sub-offset (contiguous K, matches B)
  const int kb   = (lane < 16) ? 0 : 16;      // B-frag K sub-offset
  const int ncol = lane & 15;

  v8f acc[8] = {};
  for (int kc = 0; kc < 4; ++kc) {            // K = 128 in chunks of 32
    const int k0 = kc * 32;
    v16bf a;
#pragma unroll
    for (int i = 0; i < 4; ++i) {
      a[2*i + 0] = sX[mrow * 128 + k0 + ka + 2*i + 0];
      a[2*i + 1] = sX[mrow * 128 + k0 + ka + 2*i + 1];
      a[2*i + 8] = sX[mrow * 128 + k0 + 8 + ka + 2*i + 0];
      a[2*i + 9] = sX[mrow * 128 + k0 + 8 + ka + 2*i + 1];
    }
#pragma unroll
    for (int t = 0; t < 8; ++t) {             // 8 N-tiles -> 128 cols
      v16bf b;
#pragma unroll
      for (int i = 0; i < 8; ++i) {
        b[2*i + 0] = sWt[(t * 16 + ncol) * 128 + k0 + kb + 2*i + 0];
        b[2*i + 1] = sWt[(t * 16 + ncol) * 128 + k0 + kb + 2*i + 1];
      }
      acc[t] = __builtin_amdgcn_wmma_f32_16x16x32_bf16(
          false, a, false, b, (short)0, acc[t], false, false);
    }
  }

  const int radd = (lane < 16) ? 0 : 8;       // C/D layout: VGPR j -> M=j / j+8
#pragma unroll
  for (int t = 0; t < 8; ++t) {
#pragma unroll
    for (int j = 0; j < 8; ++j) {
      const int gr = row0 + m0 + radd + j;
      if (gr < nrows) Y[(size_t)gr * 128 + t * 16 + ncol] = acc[t][j];
    }
  }
}

// ------------------ edge scatter: agg[dst] += h[src]*dinv[s]*dinv[d] --------

__global__ __launch_bounds__(256) void k_scatter(
    const float* __restrict__ H, const float* __restrict__ dinv,
    const int* __restrict__ src, const int* __restrict__ dst,
    float* __restrict__ Agg, int nedges) {
  const int gt   = blockIdx.x * 256 + threadIdx.x;
  const int e    = gt >> 5;
  const int lane = gt & 31;
  if (e >= nedges) return;
  const int s = src[e];
  const int d = dst[e];
  const float nrm = dinv[s] * dinv[d];
  const float4 h = *(const float4*)(H + (size_t)s * 128 + lane * 4);
  float* o = Agg + (size_t)d * 128 + lane * 4;
  atomicAdd(o + 0, h.x * nrm);
  atomicAdd(o + 1, h.y * nrm);
  atomicAdd(o + 2, h.z * nrm);
  atomicAdd(o + 3, h.w * nrm);
}

// ------- epilogue: Agg = relu(Agg + h*dinv^2 (self loop) + bias), in place --

__global__ __launch_bounds__(256) void k_epilogue(
    const float* __restrict__ H, const float* __restrict__ dinv,
    const float* __restrict__ bias, float* __restrict__ Agg, int n) {
  const int gt   = blockIdx.x * 256 + threadIdx.x;
  const int i    = gt >> 5;
  const int lane = gt & 31;
  if (i >= n) return;
  const float di = dinv[i];
  const float sl = di * di;
  const int f = lane * 4;
  const float4 h = *(const float4*)(H   + (size_t)i * 128 + f);
  const float4 a = *(const float4*)(Agg + (size_t)i * 128 + f);
  const float4 b = *(const float4*)(bias + f);
  float4 r;
  r.x = fmaxf(a.x + h.x * sl + b.x, 0.f);
  r.y = fmaxf(a.y + h.y * sl + b.y, 0.f);
  r.z = fmaxf(a.z + h.z * sl + b.z, 0.f);
  r.w = fmaxf(a.w + h.w * sl + b.w, 0.f);
  *(float4*)(Agg + (size_t)i * 128 + f) = r;
}

// --------- classifier: Y[N,40] = (out1+out2)[N,128] @ Wc[128,40] + bc -------

__global__ __launch_bounds__(256) void k_gcn_classifier(
    const float* __restrict__ A1, const float* __restrict__ A2,
    const float* __restrict__ Wc, const float* __restrict__ bc,
    float* __restrict__ Y, int nrows) {
  __shared__ __bf16 sX [128 * 128];
  __shared__ __bf16 sWt[48 * 128];            // [col][k], cols 40..47 zero pad
  __shared__ float  sB[48];
  const int tid  = threadIdx.x;
  const int row0 = blockIdx.x * 128;

  for (int idx = tid; idx < 48 * 128; idx += 256) {
    const int c = idx >> 7, k = idx & 127;
    sWt[idx] = (c < 40) ? f2bf(Wc[k * 40 + c]) : f2bf(0.f);
  }
  if (tid < 48) sB[tid] = (tid < 40) ? bc[tid] : 0.f;

  for (int idx = tid * 4; idx < 128 * 128; idx += 256 * 4) {
    const int r = idx >> 7, c = idx & 127;
    const int gr = row0 + r;
    float4 v = make_float4(0.f, 0.f, 0.f, 0.f);
    if (gr < nrows) {
      const float4 u = *(const float4*)(A1 + (size_t)gr * 128 + c);
      const float4 w = *(const float4*)(A2 + (size_t)gr * 128 + c);
      v.x = u.x + w.x; v.y = u.y + w.y; v.z = u.z + w.z; v.w = u.w + w.w;
    }
    sX[idx + 0] = f2bf(v.x);
    sX[idx + 1] = f2bf(v.y);
    sX[idx + 2] = f2bf(v.z);
    sX[idx + 3] = f2bf(v.w);
  }
  __syncthreads();

  const int wave = tid >> 5, lane = tid & 31;
  const int m0   = wave * 16;
  const int mrow = m0 + (lane & 15);
  const int ka   = (lane < 16) ? 0 : 16;
  const int kb   = (lane < 16) ? 0 : 16;
  const int ncol = lane & 15;

  v8f acc[3] = {};
  for (int kc = 0; kc < 4; ++kc) {
    const int k0 = kc * 32;
    v16bf a;
#pragma unroll
    for (int i = 0; i < 4; ++i) {
      a[2*i + 0] = sX[mrow * 128 + k0 + ka + 2*i + 0];
      a[2*i + 1] = sX[mrow * 128 + k0 + ka + 2*i + 1];
      a[2*i + 8] = sX[mrow * 128 + k0 + 8 + ka + 2*i + 0];
      a[2*i + 9] = sX[mrow * 128 + k0 + 8 + ka + 2*i + 1];
    }
#pragma unroll
    for (int t = 0; t < 3; ++t) {             // 48 padded cols
      v16bf b;
#pragma unroll
      for (int i = 0; i < 8; ++i) {
        b[2*i + 0] = sWt[(t * 16 + ncol) * 128 + k0 + kb + 2*i + 0];
        b[2*i + 1] = sWt[(t * 16 + ncol) * 128 + k0 + kb + 2*i + 1];
      }
      acc[t] = __builtin_amdgcn_wmma_f32_16x16x32_bf16(
          false, a, false, b, (short)0, acc[t], false, false);
    }
  }

  const int radd = (lane < 16) ? 0 : 8;
#pragma unroll
  for (int t = 0; t < 3; ++t) {
    const int col = t * 16 + ncol;
#pragma unroll
    for (int j = 0; j < 8; ++j) {
      const int gr = row0 + m0 + radd + j;
      if (gr < nrows && col < 40)
        Y[(size_t)gr * 40 + col] = acc[t][j] + sB[col];
    }
  }
}

// ------------------------------- launcher -----------------------------------

extern "C" void kernel_launch(void* const* d_in, const int* in_sizes, int n_in,
                              void* d_out, int out_size, void* d_ws, size_t ws_size,
                              hipStream_t stream) {
  const float* x  = (const float*)d_in[0];
  const float* W1 = (const float*)d_in[1];
  const float* b1 = (const float*)d_in[2];
  const float* W2 = (const float*)d_in[3];
  const float* b2 = (const float*)d_in[4];
  const float* Wc = (const float*)d_in[5];
  const float* bc = (const float*)d_in[6];
  const int*   ei = (const int*)d_in[7];

  const int n = in_sizes[0] / 128;            // nodes
  const int e = in_sizes[7] / 2;              // edges
  const int* src = ei;
  const int* dst = ei + e;

  // workspace layout (floats): deg | dinv | h | agg1(out1) | agg2(out2)
  float* ws   = (float*)d_ws;
  float* deg  = ws;
  float* dinv = ws + n;
  float* h    = dinv + n;
  float* agg1 = h    + (size_t)n * 128;
  float* agg2 = agg1 + (size_t)n * 128;
  float* out  = (float*)d_out;

  const int big4 = (n * 128) / 4;
  const dim3 blk(256);

  // re-init scratch every call (harness does not re-poison between replays)
  k_zero_f4<<<dim3((n / 4 + 255) / 256), blk, 0, stream>>>((float4*)deg, n / 4);
  k_zero_f4<<<dim3((big4 + 255) / 256), blk, 0, stream>>>((float4*)agg1, big4);
  k_zero_f4<<<dim3((big4 + 255) / 256), blk, 0, stream>>>((float4*)agg2, big4);

  // degrees and symmetric normalization
  k_deg <<<dim3((e + 255) / 256), blk, 0, stream>>>(dst, deg, e);
  k_dinv<<<dim3((n + 255) / 256), blk, 0, stream>>>(deg, dinv, n);

  const dim3 ggrid((n + 127) / 128);
  const dim3 sgrid((e * 32 + 255) / 256);
  const dim3 egrid((n * 32 + 255) / 256);

  // layer 1: h = x@W1 ; agg1 = scatter(h) ; out1 = relu(agg1 + selfloop + b1)
  k_gcn_gemm128<<<ggrid, blk, 0, stream>>>(x, W1, h, n);
  k_scatter    <<<sgrid, blk, 0, stream>>>(h, dinv, src, dst, agg1, e);
  k_epilogue   <<<egrid, blk, 0, stream>>>(h, dinv, b1, agg1, n);

  // layer 2 (reads out1 = agg1)
  k_gcn_gemm128<<<ggrid, blk, 0, stream>>>(agg1, W2, h, n);
  k_scatter    <<<sgrid, blk, 0, stream>>>(h, dinv, src, dst, agg2, e);
  k_epilogue   <<<egrid, blk, 0, stream>>>(h, dinv, b2, agg2, n);

  // classifier: out = (out2 + out1) @ Wc + bc
  k_gcn_classifier<<<ggrid, blk, 0, stream>>>(agg1, agg2, Wc, bc, out, n);
}
